// RowRNN_11819749999151
// MI455X (gfx1250) — compile-verified, matrix operand order checked by reading, not run on hardware
//
#include <hip/hip_runtime.h>

// ---------------------------------------------------------------------------
// RowRNN (3-layer column-GRU over image rows) for MI455X / gfx1250.
//  * All hot GEMMs folded to K=512 -> v_wmma_f32_16x16x32_bf16.
//  * Column scan (the serial critical path: 192 scans x 64 steps) is split
//    across NW=8 cooperating workgroups synchronized per step via L2 atomics;
//    each wave keeps its 48 Whh fragments resident in VGPRs (step-invariant),
//    so a step costs only the 32KB h exchange + 96 WMMA issues per wave.
//  * Parallel xproj / logits GEMMs are wide WMMA kernels.
// Requires ws_size >= ~31 MB.
// ---------------------------------------------------------------------------

#define B_  32
#define R_  64
#define C_  64
#define E_  256
#define H_  512
#define G3_ 1536   // 3*H
#define V_  258
#define VP_ 272    // V padded to 16
#define HPAD 520   // LDS row pitch (bf16 elems) -> conflict-free tile reads
#define NW_ 8      // workgroups cooperating on one scan
#define WAVES_ 4   // waves per scan workgroup (NW_*WAVES_ == 32 wave slots)

typedef __bf16 bf_t;
typedef __attribute__((ext_vector_type(16))) __bf16 v16bf;
typedef __attribute__((ext_vector_type(8)))  __bf16 v8bf;
typedef __attribute__((ext_vector_type(8)))  float  v8f;

static __device__ __forceinline__ float bf2f(bf_t x) {
  unsigned short s = __builtin_bit_cast(unsigned short, x);
  unsigned u = ((unsigned)s) << 16;
  return __builtin_bit_cast(float, u);
}
static __device__ __forceinline__ bf_t f2bf(float f) {
  unsigned u = __builtin_bit_cast(unsigned, f);
  unsigned r = u + 0x7FFFu + ((u >> 16) & 1u);   // round-to-nearest-even
  unsigned short s = (unsigned short)(r >> 16);
  return __builtin_bit_cast(bf_t, s);
}
static __device__ __forceinline__ bf_t bf_zero() {
  return __builtin_bit_cast(bf_t, (unsigned short)0);
}

// One 16-lane-half fragment row: 8 bf16 at p, 8 bf16 at p+16.
// CDNA5 16-bit A/B layout: lane<16 holds K {0..7,16..23}, lane>=16 holds
// K {8..15,24..31} (caller applies the +8 base via `half*8`).
static __device__ __forceinline__ v16bf frag16(const bf_t* p) {
  v8bf lo = *(const v8bf*)(p);
  v8bf hi = *(const v8bf*)(p + 16);
  return __builtin_shufflevector(lo, hi, 0,1,2,3,4,5,6,7,8,9,10,11,12,13,14,15);
}

static __device__ __forceinline__ v8f wmma_bf16(v16bf a, v16bf b, v8f c) {
  return __builtin_amdgcn_wmma_f32_16x16x32_bf16(false, a, false, b,
                                                 (short)0, c, false, false);
}

static __device__ __forceinline__ float sigmoidf_(float x) {
  return 1.0f / (1.0f + __expf(-x));
}

// ---------------------------------------------------------------------------
// Setup kernels (once per launch; negligible vs. 1.1 TFLOP of GEMMs)
// ---------------------------------------------------------------------------
__global__ void k_convert(const float* __restrict__ Whh,
                          const float* __restrict__ Wih12,
                          const float* __restrict__ outw,
                          bf_t* __restrict__ Whh_b,
                          bf_t* __restrict__ Wih12_b,
                          bf_t* __restrict__ outw_b) {
  int i = blockIdx.x * 256 + threadIdx.x;
  if (i < 3 * G3_ * H_)  Whh_b[i]   = f2bf(Whh[i]);
  if (i < 2 * G3_ * H_)  Wih12_b[i] = f2bf(Wih12[i]);
  if (i < VP_ * H_)      outw_b[i]  = (i < V_ * H_) ? f2bf(outw[i]) : bf_zero();
}

// W_hx[j][k] = sum_e Wih0[j][e] * hte_w[e][k]   (fold hidden->embed->ih0)
__global__ void k_whx(const float* __restrict__ Wih0,
                      const float* __restrict__ htew,
                      bf_t* __restrict__ Whx) {
  int idx = blockIdx.x * 256 + threadIdx.x;
  if (idx >= G3_ * H_) return;
  int j = idx >> 9, k = idx & (H_ - 1);
  float s = 0.f;
  for (int e = 0; e < E_ + 1; ++e) s += Wih0[j * (E_ + 1) + e] * htew[e * H_ + k];
  Whx[idx] = f2bf(s);
}

// emb_proj[v][j] = sum_{e<E} emb[v][e] * Wih0[j][e]
__global__ void k_embp(const float* __restrict__ emb,
                       const float* __restrict__ Wih0,
                       float* __restrict__ embp) {
  int idx = blockIdx.x * 256 + threadIdx.x;
  if (idx >= V_ * G3_) return;
  int v = idx / G3_, j = idx - v * G3_;
  float s = 0.f;
  for (int e = 0; e < E_; ++e) s += emb[v * E_ + e] * Wih0[j * (E_ + 1) + e];
  embp[idx] = s;
}

// b0f[j] = Wih0[j].hte_b + bih0[j] ; roww[j] = Wih0[j][E]
__global__ void k_bias0(const float* __restrict__ Wih0,
                        const float* __restrict__ hteb,
                        const float* __restrict__ bih,
                        float* __restrict__ b0f,
                        float* __restrict__ roww) {
  int j = blockIdx.x * 256 + threadIdx.x;
  if (j >= G3_) return;
  float s = bih[j];
  for (int e = 0; e < E_ + 1; ++e) s += Wih0[j * (E_ + 1) + e] * hteb[e];
  b0f[j]  = s;
  roww[j] = Wih0[j * (E_ + 1) + E_];
}

__global__ void k_zero_bf(bf_t* __restrict__ p, int n) {
  int i = blockIdx.x * 256 + threadIdx.x;
  if (i < n) p[i] = bf_zero();
}
__global__ void k_zero_i32(int* __restrict__ p, int n) {
  int i = blockIdx.x * 256 + threadIdx.x;
  if (i < n) p[i] = 0;
}

// ---------------------------------------------------------------------------
// Generic WMMA GEMM: Y[m][j] = sum_k A[m][k]*W[j][k] + epilogue
//   mode 0: xproj layer0 (emb gather + row coord + fused bias), Y fp32 [m][1536]
//   mode 1: xproj layer1/2 (+bih),                              Y fp32 [m][1536]
//   mode 2: output logits (+out_b), scatter to d_out [B][R][C][V]
// m index = c*32 + b.
// ---------------------------------------------------------------------------
__global__ __launch_bounds__(256) void k_gemm(
    const bf_t* __restrict__ A, const bf_t* __restrict__ W,
    const float* __restrict__ bias, float* __restrict__ Y,
    int Mtiles, int Ntiles, int Nlogical, int mode,
    const int* __restrict__ xtok, int r, float rowval,
    const float* __restrict__ embp, const float* __restrict__ roww,
    float* __restrict__ outp) {
  int wid = blockIdx.x * 8 + (threadIdx.x >> 5);
  if (wid >= Mtiles * Ntiles) return;
  int mt = wid / Ntiles, nt = wid - mt * Ntiles;
  int lane = threadIdx.x & 31;
  int ncol = lane & 15, half = lane >> 4;

  const bf_t* arow = A + ((size_t)(mt * 16 + ncol)) * H_ + half * 8;
  const bf_t* wrow = W + ((size_t)(nt * 16 + ncol)) * H_ + half * 8;

  v8f acc = {};
#pragma unroll
  for (int kt = 0; kt < H_ / 32; ++kt) {
    v16bf a = frag16(arow + kt * 32);
    v16bf b = frag16(wrow + kt * 32);
    acc = wmma_bf16(a, b, acc);
  }

  int j = nt * 16 + ncol;
  if (mode == 2) {
    if (j < Nlogical) {
      float bj = bias[j];
#pragma unroll
      for (int v = 0; v < 8; ++v) {
        int m = mt * 16 + half * 8 + v;
        int b_ = m & (B_ - 1), c_ = m >> 5;
        outp[(((size_t)b_ * R_ + r) * C_ + c_) * V_ + j] = acc[v] + bj;
      }
    }
    return;
  }
  float extra = bias[j];
  if (mode == 0) extra += rowval * roww[j];
#pragma unroll
  for (int v = 0; v < 8; ++v) {
    int m = mt * 16 + half * 8 + v;
    float val = acc[v] + extra;
    if (mode == 0) {
      int b_ = m & (B_ - 1), c_ = m >> 5;
      int tok = xtok[b_ * (R_ * C_) + r * C_ + c_];
      val += embp[(size_t)tok * G3_ + j];
    }
    Y[(size_t)m * G3_ + j] = val;
  }
}

// ---------------------------------------------------------------------------
// Cooperative column-GRU scan for one (row, layer): 64 sequential steps,
// split across NW_ workgroups (each owns 512/NW_ gate columns for all 3
// gates). Wave slot s = wg*WAVES_+wave owns cols [s*16, s*16+16):
//   * 48 Whh fragments preloaded to VGPRs (constant over all 64 steps)
//   * h exchanged via ping-pong L2 buffer + per-step arrive counter
//   * gate math entirely in registers (6 f32 WMMA accumulators per wave)
// ---------------------------------------------------------------------------
__global__ __launch_bounds__(WAVES_ * 32) void k_gru_scan(
    const bf_t* __restrict__ Whh,   // [1536][512] bf16 (this layer)
    const float* __restrict__ bhh,  // [1536]      (this layer)
    const float* __restrict__ X,    // [64][32][1536] fp32 (xproj this row)
    bf_t* __restrict__ Hout,        // [64][32][512] bf16
    bf_t* __restrict__ hshare,      // [2][32][512] bf16 ping-pong
    int* __restrict__ counters) {   // [64] this scan's step counters (zeroed)
  __shared__ bf_t h[B_][HPAD];

  const int t = threadIdx.x;
  const int lane = t & 31, wave = t >> 5;
  const int slot = blockIdx.x * WAVES_ + wave;   // 0..31
  const int ncol = lane & 15, half = lane >> 4;
  const int g0 = slot * 16 + ncol;               // gate column in [0,512)

  // h(step0) = 0
  for (int i = t; i < B_ * HPAD; i += WAVES_ * 32) (&h[0][0])[i] = bf_zero();

  // Preload this wave's Whh fragments: 3 gates x 16 k-tiles (384 VGPRs).
  v16bf Bf[3][16];
#pragma unroll
  for (int g = 0; g < 3; ++g) {
    const bf_t* wrow = Whh + ((size_t)(g * H_ + g0)) * H_ + half * 8;
#pragma unroll
    for (int kt = 0; kt < 16; ++kt) Bf[g][kt] = frag16(wrow + kt * 32);
  }
  const float bR = bhh[g0], bZ = bhh[H_ + g0], bN = bhh[2 * H_ + g0];
  __syncthreads();

  for (int s = 0; s < C_; ++s) {
    if (s > 0) {
      // wait for all workgroups to publish h(s-1), then pull it into LDS
      if (t == 0) {
        while (__hip_atomic_load(&counters[s - 1], __ATOMIC_ACQUIRE,
                                 __HIP_MEMORY_SCOPE_AGENT) < NW_)
          __builtin_amdgcn_s_sleep(1);
      }
      __syncthreads();
      const bf_t* src = hshare + ((size_t)((s + 1) & 1)) * B_ * H_;
      for (int cid = t; cid < B_ * H_ / 8; cid += WAVES_ * 32) {
        int row = cid >> 6, c8 = cid & 63;
        *(v8bf*)&h[row][c8 * 8] = *(const v8bf*)(src + row * H_ + c8 * 8);
      }
      __syncthreads();
    }

    v8f a00 = {}, a01 = {}, a02 = {}, a10 = {}, a11 = {}, a12 = {};
#pragma unroll
    for (int kt = 0; kt < 16; ++kt) {
      int ko = kt * 32 + half * 8;
      v16bf fa0 = frag16(&h[ncol][ko]);        // M rows 0..15
      v16bf fa1 = frag16(&h[16 + ncol][ko]);   // M rows 16..31
      a00 = wmma_bf16(fa0, Bf[0][kt], a00);  a10 = wmma_bf16(fa1, Bf[0][kt], a10);
      a01 = wmma_bf16(fa0, Bf[1][kt], a01);  a11 = wmma_bf16(fa1, Bf[1][kt], a11);
      a02 = wmma_bf16(fa0, Bf[2][kt], a02);  a12 = wmma_bf16(fa1, Bf[2][kt], a12);
    }

    bf_t* dst = hshare + ((size_t)(s & 1)) * B_ * H_;
    const float* xbase = X + (size_t)s * B_ * G3_;
#pragma unroll
    for (int m = 0; m < 2; ++m) {
#pragma unroll
      for (int v = 0; v < 8; ++v) {
        int row = m * 16 + half * 8 + v;
        const float* xp = xbase + (size_t)row * G3_;
        float hr = (m ? a10[v] : a00[v]) + bR;
        float hz = (m ? a11[v] : a01[v]) + bZ;
        float hn = (m ? a12[v] : a02[v]) + bN;
        float rr = sigmoidf_(xp[g0] + hr);
        float zz = sigmoidf_(xp[H_ + g0] + hz);
        float nn = tanhf(xp[2 * H_ + g0] + rr * hn);
        float ho = bf2f(h[row][g0]);
        float hv = (1.f - zz) * nn + zz * ho;
        bf_t hb = f2bf(hv);
        dst[(size_t)row * H_ + g0] = hb;
        Hout[((size_t)s * B_ + row) * H_ + g0] = hb;
      }
    }

    // publish h(s): make slice writes visible, then arrive
    __threadfence();
    __syncthreads();
    if (t == 0)
      __hip_atomic_fetch_add(&counters[s], 1, __ATOMIC_RELEASE,
                             __HIP_MEMORY_SCOPE_AGENT);
  }
}

// ---------------------------------------------------------------------------
extern "C" void kernel_launch(void* const* d_in, const int* in_sizes, int n_in,
                              void* d_out, int out_size, void* d_ws, size_t ws_size,
                              hipStream_t stream) {
  const int*   x     = (const int*)d_in[0];
  const float* emb   = (const float*)d_in[1];
  const float* Wih0  = (const float*)d_in[2];
  const float* Wih12 = (const float*)d_in[3];
  const float* Whh   = (const float*)d_in[4];
  const float* bih   = (const float*)d_in[5];
  const float* bhh   = (const float*)d_in[6];
  const float* htew  = (const float*)d_in[7];
  const float* hteb  = (const float*)d_in[8];
  const float* outw  = (const float*)d_in[9];
  const float* outb  = (const float*)d_in[10];
  float* out = (float*)d_out;

  char* ws = (char*)d_ws;
  auto alloc = [&](size_t bytes) -> char* {
    char* p = ws; ws += (bytes + 255) & ~(size_t)255; return p;
  };
  bf_t*  Whh_b   = (bf_t*) alloc((size_t)3 * G3_ * H_ * 2);
  bf_t*  Wih12_b = (bf_t*) alloc((size_t)2 * G3_ * H_ * 2);
  bf_t*  Whx_b   = (bf_t*) alloc((size_t)G3_ * H_ * 2);
  bf_t*  outw_b  = (bf_t*) alloc((size_t)VP_ * H_ * 2);
  float* embp    = (float*)alloc((size_t)V_ * G3_ * 4);
  float* b0f     = (float*)alloc((size_t)G3_ * 4);
  float* roww    = (float*)alloc((size_t)G3_ * 4);
  float* X       = (float*)alloc((size_t)C_ * B_ * G3_ * 4);
  bf_t*  Hprev   = (bf_t*) alloc((size_t)C_ * B_ * H_ * 2);
  bf_t*  Ha      = (bf_t*) alloc((size_t)C_ * B_ * H_ * 2);
  bf_t*  Hb      = (bf_t*) alloc((size_t)C_ * B_ * H_ * 2);
  bf_t*  hshare  = (bf_t*) alloc((size_t)2 * B_ * H_ * 2);
  int*   counters= (int*)  alloc((size_t)R_ * 3 * C_ * 4);   // 192 scans x 64

  // ---- one-time (per launch) weight folding + state init ----
  {
    int n = 3 * G3_ * H_;
    k_convert<<<dim3((n + 255) / 256), dim3(256), 0, stream>>>(
        Whh, Wih12, outw, Whh_b, Wih12_b, outw_b);
    n = G3_ * H_;
    k_whx<<<dim3((n + 255) / 256), dim3(256), 0, stream>>>(Wih0, htew, Whx_b);
    n = V_ * G3_;
    k_embp<<<dim3((n + 255) / 256), dim3(256), 0, stream>>>(emb, Wih0, embp);
    k_bias0<<<dim3((G3_ + 255) / 256), dim3(256), 0, stream>>>(
        Wih0, hteb, bih, b0f, roww);
    n = C_ * B_ * H_;
    k_zero_bf<<<dim3((n + 255) / 256), dim3(256), 0, stream>>>(Hprev, n);
    n = R_ * 3 * C_;
    k_zero_i32<<<dim3((n + 255) / 256), dim3(256), 0, stream>>>(counters, n);
  }

  const int Mtiles  = (C_ * B_) / 16;           // 128
  const int NtilesG = G3_ / 16;                 // 96
  const int NtilesV = VP_ / 16;                 // 17
  dim3 gridG(((Mtiles * NtilesG) + 7) / 8);     // 1536 blocks of 8 waves
  dim3 gridV(((Mtiles * NtilesV) + 7) / 8);     // 272 blocks
  dim3 gridS(NW_), blkS(WAVES_ * 32);

  for (int r = 0; r < R_; ++r) {
    float rv = ((float)r / (float)R_) * 2.f - 1.f;
    int si = r * 3;

    // layer 0 xproj: X = emb_proj[tok] + rv*roww + b0f + Hprev @ Whx^T
    k_gemm<<<gridG, dim3(256), 0, stream>>>(
        Hprev, Whx_b, b0f, X, Mtiles, NtilesG, G3_, 0,
        x, r, rv, embp, roww, nullptr);
    k_gru_scan<<<gridS, blkS, 0, stream>>>(
        Whh_b, bhh, X, Ha, hshare, counters + (size_t)(si + 0) * C_);

    // layer 1
    k_gemm<<<gridG, dim3(256), 0, stream>>>(
        Ha, Wih12_b, bih + G3_, X, Mtiles, NtilesG, G3_, 1,
        nullptr, r, 0.f, nullptr, nullptr, nullptr);
    k_gru_scan<<<gridS, blkS, 0, stream>>>(
        Whh_b + (size_t)G3_ * H_, bhh + G3_, X, Hb,
        hshare, counters + (size_t)(si + 1) * C_);

    // layer 2 (output becomes prev_h for next row)
    k_gemm<<<gridG, dim3(256), 0, stream>>>(
        Hb, Wih12_b + (size_t)G3_ * H_, bih + 2 * G3_, X, Mtiles, NtilesG, G3_, 1,
        nullptr, r, 0.f, nullptr, nullptr, nullptr);
    k_gru_scan<<<gridS, blkS, 0, stream>>>(
        Whh_b + (size_t)2 * G3_ * H_, bhh + 2 * G3_, X, Hprev,
        hshare, counters + (size_t)(si + 2) * C_);

    // logits for this row
    k_gemm<<<gridV, dim3(256), 0, stream>>>(
        Hprev, outw_b, outb, nullptr, Mtiles, NtilesV, V_, 2,
        nullptr, r, 0.f, nullptr, nullptr, out);
  }
}
